// Discriminator_16612933501585
// MI455X (gfx1250) — compile-verified, matrix operand order checked by reading, not run on hardware
//
#include <hip/hip_runtime.h>

// ---------------------------------------------------------------------------
// Discriminator fused pipeline for gfx1250 (MI455X), bf16 WMMA everywhere hot.
// B=16, N=64, F=512, IMF=2048.
// ---------------------------------------------------------------------------

typedef __bf16 bf16_t;
typedef bf16_t bfx8  __attribute__((ext_vector_type(8)));
typedef bf16_t bfx16 __attribute__((ext_vector_type(16)));
typedef float  fx8   __attribute__((ext_vector_type(8)));
typedef unsigned int u32x4 __attribute__((ext_vector_type(4)));
typedef unsigned int u32x8 __attribute__((ext_vector_type(8)));

#define WMMA_BF16(a, b, c) \
    __builtin_amdgcn_wmma_f32_16x16x32_bf16(false, (a), false, (b), (short)0, (c), false, false)

#if defined(__HIP_DEVICE_COMPILE__) && defined(__gfx1250__)
#define CDNA5_ASM 1
#else
#define CDNA5_ASM 0
#endif

// Build a 16x32 bf16 A/B fragment from a row-major source. Per §7.12.2:
// lanes 0-15 take K = [k..k+7] and [k+16..k+23]; lanes 16-31 take +8 on both.
// Caller passes p = base + row*ld + k + half*8. Two 16B chunks -> ds/global b128.
__device__ __forceinline__ bfx16 make_frag(const bf16_t* p) {
    bfx8 lo = *(const bfx8*)(p);
    bfx8 hi = *(const bfx8*)(p + 16);
    bfx16 r;
#pragma unroll
    for (int t = 0; t < 8; ++t) { r[t] = lo[t]; r[t + 8] = hi[t]; }
    return r;
}

// ---------------------------------------------------------------------------
// Prep: f32 -> bf16 weight/activation conversion, Wc column-block sums,
// zero the sent_sum accumulator. One flat grid, section-dispatched.
// ---------------------------------------------------------------------------
__global__ void prep_kernel(const float* __restrict__ input,
                            const float* __restrict__ g1w,
                            const float* __restrict__ g2w,
                            const float* __restrict__ g3w,
                            const float* __restrict__ im_in,
                            const float* __restrict__ im_w,
                            bf16_t* __restrict__ inp_bf,
                            bf16_t* __restrict__ w1a,
                            float*  __restrict__ wc,
                            bf16_t* __restrict__ w2,
                            bf16_t* __restrict__ w3,
                            bf16_t* __restrict__ imin_bf,
                            bf16_t* __restrict__ imw_bf,
                            float*  __restrict__ sent_sum) {
    long e = (long)blockIdx.x * 256 + threadIdx.x;
    if (e < 524288) { inp_bf[e] = (bf16_t)input[e]; return; }           // input [16,64,512]
    e -= 524288;
    if (e < 262144) {                                                   // W1a = g1w[:, :512]
        long o = e >> 9, k = e & 511;
        w1a[e] = (bf16_t)g1w[o * 1024 + k];
        return;
    }
    e -= 262144;
    if (e < 4096) {                                                     // Wc[o,q] = sum_j g1w[o, 512+64q+j]
        long o = e >> 3, q = e & 7;
        const float* p = g1w + o * 1024 + 512 + q * 64;
        float s = 0.f;
#pragma unroll 8
        for (int j = 0; j < 64; ++j) s += p[j];
        wc[e] = s;
        return;
    }
    e -= 4096;
    if (e < 262144) { w2[e] = (bf16_t)g2w[e]; return; }
    e -= 262144;
    if (e < 262144) { w3[e] = (bf16_t)g3w[e]; return; }
    e -= 262144;
    if (e < 32768)  { imin_bf[e] = (bf16_t)im_in[e]; return; }          // im_input [16,2048]
    e -= 32768;
    if (e < 1048576){ imw_bf[e] = (bf16_t)im_w[e]; return; }            // im_w [512,2048]
    e -= 1048576;
    if (e < 8192)   { sent_sum[e] = 0.f; return; }                      // [16,512]
}

// ---------------------------------------------------------------------------
// X1[b] = input[b] @ W1a^T   -> [16,64,512] f32.  One block per batch.
// ---------------------------------------------------------------------------
__global__ __launch_bounds__(256) void x1_kernel(const bf16_t* __restrict__ inp_bf,
                                                 const bf16_t* __restrict__ w1a,
                                                 float* __restrict__ X1) {
    const int b    = blockIdx.x;
    const int tid  = threadIdx.x;
    const int wv   = tid >> 5;
    const int lane = tid & 31;
    const int l    = lane & 15;
    const int half = lane >> 4;
    const bf16_t* A = inp_bf + (size_t)b * 64 * 512;

    for (int t = wv; t < 128; t += 8) {            // 4 Mtiles x 32 Ntiles
        const int mt = t >> 5, nt = t & 31;
        fx8 acc = {};
        for (int k = 0; k < 512; k += 32) {
            bfx16 af = make_frag(A   + (mt * 16 + l) * 512 + k + half * 8);
            bfx16 bf = make_frag(w1a + (nt * 16 + l) * 512 + k + half * 8);
            acc = WMMA_BF16(af, bf, acc);
        }
#pragma unroll
        for (int r = 0; r < 8; ++r) {
            const int row = mt * 16 + half * 8 + r;
            const int col = nt * 16 + l;
            X1[((size_t)b * 64 + row) * 512 + col] = acc[r];
        }
    }
}

// ---------------------------------------------------------------------------
// img_pre = im_input @ im_w^T + im_b  -> [16,512] f32.  Single block, M=16.
// ---------------------------------------------------------------------------
__global__ __launch_bounds__(256) void im_kernel(const bf16_t* __restrict__ imin_bf,
                                                 const bf16_t* __restrict__ imw_bf,
                                                 const float* __restrict__ im_b,
                                                 float* __restrict__ img_pre) {
    const int tid  = threadIdx.x;
    const int wv   = tid >> 5;
    const int lane = tid & 31;
    const int l    = lane & 15;
    const int half = lane >> 4;

    for (int nt = wv; nt < 32; nt += 8) {
        fx8 acc = {};
        for (int k = 0; k < 2048; k += 32) {
            bfx16 af = make_frag(imin_bf + l * 2048 + k + half * 8);
            bfx16 bf = make_frag(imw_bf + (nt * 16 + l) * 2048 + k + half * 8);
            acc = WMMA_BF16(af, bf, acc);
        }
        const int col  = nt * 16 + l;
        const float bb = im_b[col];
#pragma unroll
        for (int r = 0; r < 8; ++r) {
            const int row = half * 8 + r;
            img_pre[row * 512 + col] = acc[r] + bb;
        }
    }
}

// ---------------------------------------------------------------------------
// Fused relation-MLP tile: one block per (i, b). h1 comes from the algebraic
// collapse of GEMM1 (X1 + Wc dot); GEMM2/GEMM3 run fully in LDS with 4x
// B-fragment register reuse; column sums accumulate into sent_sum.
// LDS: inrow 2KB + Wc 16KB + h1 65KB + h2 65KB (~148KB).
// Row stride 520 (bf16) => 260 dwords == 4 mod 64 banks: conflict-free b128.
// Data movement: Wc staged via TDM tensor_load_to_lds (TENSORcnt), inrow via
// global_load_async_to_lds_b32 (ASYNCcnt).
// ---------------------------------------------------------------------------
#define LDH 520

__global__ __launch_bounds__(256) void fused_kernel(const float* __restrict__ input,
                                                    const float* __restrict__ X1,
                                                    const float* __restrict__ wc,
                                                    const float* __restrict__ b1,
                                                    const bf16_t* __restrict__ W2,
                                                    const float* __restrict__ b2,
                                                    const bf16_t* __restrict__ W3,
                                                    const float* __restrict__ b3,
                                                    float* __restrict__ sent_sum) {
    __shared__ float  inrow[512];
    __shared__ float  wcs[4096];
    __shared__ bf16_t h1[64 * LDH];
    __shared__ bf16_t h2[64 * LDH];

    const int i = blockIdx.x;   // first pair index
    const int b = blockIdx.y;   // batch
    const int tid = threadIdx.x;
    const int wv   = tid >> 5;
    const int lane = tid & 31;
    const int l    = lane & 15;
    const int half = lane >> 4;

#if CDNA5_ASM
    // --- Wc [512,8] f32 (16KB) via Tensor Data Mover: 1D tile, one descriptor.
    if (wv == 0) {
        const unsigned long long ga = (unsigned long long)(const void*)wc;
        const unsigned ldso = (unsigned)(unsigned long long)(void*)&wcs[0];
        u32x4 g0;
        g0[0] = 1u;                                   // count=1, user mode
        g0[1] = ldso;                                 // lds_addr
        g0[2] = (unsigned)ga;                         // global_addr[31:0]
        g0[3] = (unsigned)((ga >> 32) & 0x01FFFFFFu)  // global_addr[56:32]
              | (2u << 30);                           // type=2 ("image")
        u32x8 g1;
        g1[0] = (2u << 16);      // workgroup_mask=0, data_size=2 (4 bytes)
        g1[1] = (4096u << 16);   // tensor_dim0[15:0]=4096 (bits 63:48)
        g1[2] = (1u << 16);      // tensor_dim0[31:16]=0 | tensor_dim1[15:0]=1
        g1[3] = (4096u << 16);   // tensor_dim1[31:16]=0 | tile_dim0=4096
        g1[4] = 1u;              // tile_dim1=1, tile_dim2=0
        g1[5] = 4096u;           // tensor_dim0_stride[31:0]=4096
        g1[6] = 0u;              // stride hi / dim1_stride lo
        g1[7] = 0u;              // dim1_stride hi
        asm volatile("tensor_load_to_lds %0, %1\n\t"
                     "s_wait_tensorcnt 0x0"
                     :: "s"(g0), "s"(g1) : "memory");
    }
    // --- input[b,i,:] (2KB) via async global->LDS copy, one b32 per lane.
    {
        const float* gsrc = input + ((size_t)b * 64 + i) * 512;
        for (int e = tid; e < 512; e += 256) {
            const unsigned ldso = (unsigned)(unsigned long long)(void*)&inrow[e];
            const float* gp = gsrc + e;
            asm volatile("global_load_async_to_lds_b32 %0, %1, off"
                         :: "v"(ldso), "v"(gp) : "memory");
        }
        asm volatile("s_wait_asynccnt 0x0" ::: "memory");
    }
#else
    for (int e = tid; e < 512; e += 256)
        inrow[e] = input[((size_t)b * 64 + i) * 512 + e];
    for (int e = tid; e < 4096; e += 256)
        wcs[e] = wc[e];
#endif
    __syncthreads();

    // h1[s,o] = relu( X1[b,s,o] + sum_q inrow[8s+q]*Wc[o,q] + b1[o] )
    for (int e = tid; e < 64 * 512; e += 256) {
        const int s = e >> 9, o = e & 511;
        const float* x = &inrow[s * 8];
        const float* w = &wcs[o * 8];
        float y = 0.f;
#pragma unroll
        for (int q = 0; q < 8; ++q) y += x[q] * w[q];
        const float v = X1[((size_t)b * 64 + s) * 512 + o] + y + b1[o];
        h1[s * LDH + o] = (bf16_t)fmaxf(v, 0.f);
    }
    __syncthreads();

    // GEMM2: h2 = relu(h1 @ W2^T + b2).  Each wave: 4 N-tiles, B fragment
    // loaded once per k-step and reused across all 4 M-tiles (4 accumulators).
    for (int nn = 0; nn < 4; ++nn) {
        const int nt = wv * 4 + nn;
        fx8 a0 = {}, a1 = {}, a2 = {}, a3 = {};
        for (int k = 0; k < 512; k += 32) {
            __builtin_prefetch(W2 + (nt * 16 + l) * 512 + k + 64, 0, 1);
            bfx16 bfr = make_frag(W2 + (nt * 16 + l) * 512 + k + half * 8);
            bfx16 f0 = make_frag(&h1[( 0 + l) * LDH + k + half * 8]);
            bfx16 f1 = make_frag(&h1[(16 + l) * LDH + k + half * 8]);
            bfx16 f2 = make_frag(&h1[(32 + l) * LDH + k + half * 8]);
            bfx16 f3 = make_frag(&h1[(48 + l) * LDH + k + half * 8]);
            a0 = WMMA_BF16(f0, bfr, a0);
            a1 = WMMA_BF16(f1, bfr, a1);
            a2 = WMMA_BF16(f2, bfr, a2);
            a3 = WMMA_BF16(f3, bfr, a3);
        }
        const int col  = nt * 16 + l;
        const float bb = b2[col];
#pragma unroll
        for (int r = 0; r < 8; ++r) {
            const int rr = half * 8 + r;
            h2[( 0 + rr) * LDH + col] = (bf16_t)fmaxf(a0[r] + bb, 0.f);
            h2[(16 + rr) * LDH + col] = (bf16_t)fmaxf(a1[r] + bb, 0.f);
            h2[(32 + rr) * LDH + col] = (bf16_t)fmaxf(a2[r] + bb, 0.f);
            h2[(48 + rr) * LDH + col] = (bf16_t)fmaxf(a3[r] + bb, 0.f);
        }
    }
    __syncthreads();

    // GEMM3 + ReLU + column-sum of all 64 rows into sent_sum.
    for (int nn = 0; nn < 4; ++nn) {
        const int nt = wv * 4 + nn;
        fx8 a0 = {}, a1 = {}, a2 = {}, a3 = {};
        for (int k = 0; k < 512; k += 32) {
            __builtin_prefetch(W3 + (nt * 16 + l) * 512 + k + 64, 0, 1);
            bfx16 bfr = make_frag(W3 + (nt * 16 + l) * 512 + k + half * 8);
            bfx16 f0 = make_frag(&h2[( 0 + l) * LDH + k + half * 8]);
            bfx16 f1 = make_frag(&h2[(16 + l) * LDH + k + half * 8]);
            bfx16 f2 = make_frag(&h2[(32 + l) * LDH + k + half * 8]);
            bfx16 f3 = make_frag(&h2[(48 + l) * LDH + k + half * 8]);
            a0 = WMMA_BF16(f0, bfr, a0);
            a1 = WMMA_BF16(f1, bfr, a1);
            a2 = WMMA_BF16(f2, bfr, a2);
            a3 = WMMA_BF16(f3, bfr, a3);
        }
        const int col  = nt * 16 + l;
        const float bb = b3[col];
        float part = 0.f;
#pragma unroll
        for (int r = 0; r < 8; ++r) {
            part += fmaxf(a0[r] + bb, 0.f);
            part += fmaxf(a1[r] + bb, 0.f);
            part += fmaxf(a2[r] + bb, 0.f);
            part += fmaxf(a3[r] + bb, 0.f);
        }
        atomicAdd(&sent_sum[b * 512 + col], part);
    }
}

// ---------------------------------------------------------------------------
// BatchNorm (training stats over B=16) + ReLU + assemble emb = [sent | img].
// ---------------------------------------------------------------------------
__global__ void bn_emb_kernel(const float* __restrict__ img_pre,
                              const float* __restrict__ gamma,
                              const float* __restrict__ beta,
                              const float* __restrict__ sent_sum,
                              float* __restrict__ emb) {
    const int o = blockIdx.x * 256 + threadIdx.x;
    if (o >= 512) return;
    float mu = 0.f;
#pragma unroll
    for (int r = 0; r < 16; ++r) mu += img_pre[r * 512 + o];
    mu *= (1.f / 16.f);
    float var = 0.f;
#pragma unroll
    for (int r = 0; r < 16; ++r) {
        const float d = img_pre[r * 512 + o] - mu;
        var += d * d;
    }
    var *= (1.f / 16.f);
    const float inv = rsqrtf(var + 1e-5f);
    const float g = gamma[o], bb = beta[o];
#pragma unroll
    for (int r = 0; r < 16; ++r) {
        const float v = g * (img_pre[r * 512 + o] - mu) * inv + bb;
        emb[r * 1024 + 512 + o] = fmaxf(v, 0.f);
        emb[r * 1024 + o]       = sent_sum[r * 512 + o] * (1.f / 4096.f);
    }
}

// net_D layer 1: hD = relu(emb @ d1_w^T + d1_b)   [16,512]
__global__ void d1_kernel(const float* __restrict__ emb,
                          const float* __restrict__ d1w,
                          const float* __restrict__ d1b,
                          float* __restrict__ hD) {
    const int e = blockIdx.x * 256 + threadIdx.x;   // < 8192
    const int r = e >> 9, o = e & 511;
    const float* x = emb + r * 1024;
    const float* w = d1w + o * 1024;
    float s = d1b[o];
    for (int k = 0; k < 1024; ++k) s += x[k] * w[k];
    hD[e] = fmaxf(s, 0.f);
}

// net_D layer 2: out = hD @ d2_w^T + d2_b   [16,2]
__global__ void d2_kernel(const float* __restrict__ hD,
                          const float* __restrict__ d2w,
                          const float* __restrict__ d2b,
                          float* __restrict__ out) {
    const int e = threadIdx.x;
    if (e >= 32) return;
    const int r = e >> 1, c = e & 1;
    const float* x = hD + r * 512;
    const float* w = d2w + c * 512;
    float s = d2b[c];
    for (int k = 0; k < 512; ++k) s += x[k] * w[k];
    out[e] = s;
}

// ---------------------------------------------------------------------------
extern "C" void kernel_launch(void* const* d_in, const int* in_sizes, int n_in,
                              void* d_out, int out_size, void* d_ws, size_t ws_size,
                              hipStream_t stream) {
    const float* input    = (const float*)d_in[0];
    const float* im_input = (const float*)d_in[1];
    const float* g1w      = (const float*)d_in[2];
    const float* g1b      = (const float*)d_in[3];
    const float* g2w      = (const float*)d_in[4];
    const float* g2b      = (const float*)d_in[5];
    const float* g3w      = (const float*)d_in[6];
    const float* g3b      = (const float*)d_in[7];
    const float* imw      = (const float*)d_in[8];
    const float* imb      = (const float*)d_in[9];
    const float* gamma    = (const float*)d_in[10];
    const float* beta     = (const float*)d_in[11];
    const float* d1w      = (const float*)d_in[12];
    const float* d1b      = (const float*)d_in[13];
    const float* d2w      = (const float*)d_in[14];
    const float* d2b      = (const float*)d_in[15];
    float* out = (float*)d_out;
    (void)in_sizes; (void)n_in; (void)out_size; (void)ws_size;

    char* ws = (char*)d_ws;
    size_t off = 0;
    auto take = [&](size_t bytes) -> char* {
        char* p = ws + off;
        off = (off + bytes + 255) & ~(size_t)255;
        return p;
    };
    bf16_t* inp_bf  = (bf16_t*)take(524288 * 2);   // input bf16
    bf16_t* w1a     = (bf16_t*)take(262144 * 2);   // W1[:, :512] bf16
    float*  wc      = (float*) take(4096 * 4);     // column-block sums of W1[:, 512:]
    bf16_t* w2      = (bf16_t*)take(262144 * 2);
    bf16_t* w3      = (bf16_t*)take(262144 * 2);
    bf16_t* imin_bf = (bf16_t*)take(32768 * 2);
    bf16_t* imw_bf  = (bf16_t*)take(1048576 * 2);
    float*  sent    = (float*) take(8192 * 4);     // [16,512] accumulator
    float*  X1      = (float*) take(524288 * 4);   // [16,64,512]
    float*  img_pre = (float*) take(8192 * 4);     // [16,512]
    float*  emb     = (float*) take(16384 * 4);    // [16,1024]
    float*  hD      = (float*) take(8192 * 4);     // [16,512]

    prep_kernel<<<9392, 256, 0, stream>>>(input, g1w, g2w, g3w, im_input, imw,
                                          inp_bf, w1a, wc, w2, w3, imin_bf, imw_bf, sent);
    x1_kernel<<<16, 256, 0, stream>>>(inp_bf, w1a, X1);
    im_kernel<<<1, 256, 0, stream>>>(imin_bf, imw_bf, imb, img_pre);
    fused_kernel<<<dim3(64, 16), 256, 0, stream>>>(input, X1, wc, g1b, w2, g2b, w3, g3b, sent);
    bn_emb_kernel<<<2, 256, 0, stream>>>(img_pre, gamma, beta, sent, emb);
    d1_kernel<<<32, 256, 0, stream>>>(emb, d1w, d1b, hD);
    d2_kernel<<<1, 32, 0, stream>>>(hD, d2w, d2b, out);
}
